// DFlashQwen3Attention_48430051230314
// MI455X (gfx1250) — matrix operand order repaired and checked by reading, not compiled, not run on hardware
//
#include <hip/hip_runtime.h>
#include <hip/hip_bf16.h>
#include <stdint.h>

typedef __bf16 bf16;
typedef __attribute__((ext_vector_type(16))) __bf16 v16bf;
typedef __attribute__((ext_vector_type(8)))  float  v8f;
typedef int v4i __attribute__((vector_size(16)));
typedef __attribute__((address_space(1))) v4i glob_v4i;
typedef __attribute__((address_space(3))) v4i lds_v4i;

#define NH   32
#define NKVH 8
#define HD   128
#define HID  4096
#define CTXL 4096
#define QLEN 2048
#define KVL  (CTXL + QLEN)
#define GRP  (NH / NKVH)
#define QKVC 6144   /* (H + 2*KVH) * D columns of w_qkv */

// ---------------------------------------------------------------- WMMA helpers

__device__ __forceinline__ v8f wmma_bf16(v16bf a, v16bf b, v8f c) {
  return __builtin_amdgcn_wmma_f32_16x16x32_bf16(false, a, false, b, (short)0, c,
                                                 false, false);
}

// A-fragment (16x32 bf16). Lane (g = lane>>4, r = lane&15) holds row r.
// Element e: K = g*16+e; contiguous b32 pairs.
__device__ __forceinline__ v16bf frag_a_row(const bf16* rowbase, int g) {
  v16bf o;
#pragma unroll
  for (int v = 0; v < 8; ++v) {
    int k = g * 16 + 2 * v;
    o[2 * v]     = rowbase[k];
    o[2 * v + 1] = rowbase[k + 1];
  }
  return o;
}

// B-fragment (32x16 bf16): lane (g, n) holds column n; element e -> K = g*16+e.
__device__ __forceinline__ v16bf frag_b16(const bf16* p) {
  v16bf o;
#pragma unroll
  for (int e = 0; e < 16; ++e) o[e] = p[e];
  return o;
}

// ---------------------------------------------------------------- async copy
// CDNA5 GLOBAL_LOAD_ASYNC_TO_LDS_B128: per-lane 16B global->LDS, ASYNCcnt.

__device__ __forceinline__ void cp16(const bf16* g, bf16* l) {
  __builtin_amdgcn_global_load_async_to_lds_b128(
      (glob_v4i*)(uintptr_t)g, (lds_v4i*)(unsigned)(uintptr_t)l, 0, 0);
}

__device__ __forceinline__ void wait_stage() {
  __builtin_amdgcn_s_wait_asynccnt(0);
}

// ---------------------------------------------------------------- reductions

__device__ __forceinline__ float wave_sum32(float x) {
#pragma unroll
  for (int m = 16; m >= 1; m >>= 1) x += __shfl_xor(x, m, 32);
  return x;
}
__device__ __forceinline__ float half_max16(float x) {
#pragma unroll
  for (int m = 8; m >= 1; m >>= 1) x = fmaxf(x, __shfl_xor(x, m, 32));
  return x;
}
__device__ __forceinline__ float half_sum16(float x) {
#pragma unroll
  for (int m = 8; m >= 1; m >>= 1) x += __shfl_xor(x, m, 32);
  return x;
}

// ---------------------------------------------------------------- cast kernels

__global__ void cast_f32_bf16(bf16* __restrict__ dst, const float* __restrict__ src,
                              long n) {
  long i = (long)blockIdx.x * blockDim.x + threadIdx.x;
  long stride = (long)gridDim.x * blockDim.x;
  for (; i < n; i += stride) dst[i] = (bf16)src[i];
}

// dst[n*rows + k] = (bf16)src[k*cols + n]  -- 32x32 LDS-tiled transpose
__global__ __launch_bounds__(256) void cast_transpose_f32_bf16(
    bf16* __restrict__ dst, const float* __restrict__ src, int rows, int cols) {
  __shared__ float tile[32][33];
  const int bx = blockIdx.x * 32;  // n
  const int by = blockIdx.y * 32;  // k
  const int tx = threadIdx.x & 31;
  const int ty = threadIdx.x >> 5;  // 0..7
#pragma unroll
  for (int i = 0; i < 32; i += 8)
    tile[ty + i][tx] = src[(long)(by + ty + i) * cols + bx + tx];
  __syncthreads();
#pragma unroll
  for (int i = 0; i < 32; i += 8)
    dst[(long)(bx + ty + i) * rows + by + tx] = (bf16)tile[tx][ty + i];
}

// ---------------------------------------------------------------- GEMM (bf16 -> f32)
// C[M,N] = A[M,K] * Bt[N,K]^T; A row-major, Bt = B pre-transposed [N][K].
// Block tile 128x256xK32, double-buffered async-to-LDS staging; 8 waves in 2x4,
// each wave a 64x64 patch = 16 wmma accumulators.

#define AST 40  // LDS row stride in bf16 (80B rows, 16B aligned)

__global__ __launch_bounds__(256) void gemm_bf16f32(
    const bf16* __restrict__ A, int lda, const bf16* __restrict__ Bt, int ldb,
    float* __restrict__ C, int ldc, int K) {
  __shared__ __attribute__((aligned(16))) bf16 As[2][128 * AST];
  __shared__ __attribute__((aligned(16))) bf16 Bs[2][256 * AST];

  const int t    = threadIdx.x;
  const int lane = t & 31;
  const int wave = t >> 5;
  const int g    = lane >> 4;
  const int ln   = lane & 15;
  const int wm   = (wave >> 2) * 64;
  const int wn   = (wave & 3) * 64;
  const int rowA0 = blockIdx.y * 128;
  const int colB0 = blockIdx.x * 256;

  const int arow  = t >> 1;
  const int akoff = (t & 1) * 16;

  v8f acc[4][4];
#pragma unroll
  for (int i = 0; i < 4; ++i)
#pragma unroll
    for (int j = 0; j < 4; ++j) acc[i][j] = (v8f){0, 0, 0, 0, 0, 0, 0, 0};

  auto stage = [&](int sb, int k0) {
    const bf16* ga = A + (size_t)(rowA0 + arow) * lda + k0 + akoff;
    bf16* la = &As[sb][arow * AST + akoff];
    cp16(ga, la);
    cp16(ga + 8, la + 8);
    const bf16* gb = Bt + (size_t)(colB0 + t) * ldb + k0;
    bf16* lb = &Bs[sb][t * AST];
#pragma unroll
    for (int q = 0; q < 4; ++q) cp16(gb + 8 * q, lb + 8 * q);
  };

  stage(0, 0);
  int buf = 0;
  for (int k0 = 0; k0 < K; k0 += 32) {
    wait_stage();
    __syncthreads();
    if (k0 + 32 < K) stage(buf ^ 1, k0 + 32);

    v16bf af[4];
#pragma unroll
    for (int i = 0; i < 4; ++i)
      af[i] = frag_a_row(&As[buf][(wm + i * 16 + ln) * AST], g);
#pragma unroll
    for (int j = 0; j < 4; ++j) {
      v16bf bfr = frag_b16(&Bs[buf][(wn + j * 16 + ln) * AST + g * 16]);
#pragma unroll
      for (int i = 0; i < 4; ++i) acc[i][j] = wmma_bf16(af[i], bfr, acc[i][j]);
    }
    __syncthreads();
    buf ^= 1;
  }

#pragma unroll
  for (int i = 0; i < 4; ++i) {
    const int row0 = rowA0 + wm + i * 16 + 8 * g;
#pragma unroll
    for (int j = 0; j < 4; ++j) {
      const int idx0 = row0 * ldc + colB0 + wn + j * 16 + ln;
#pragma unroll
      for (int r = 0; r < 8; ++r) C[idx0 + r * ldc] = acc[i][j][r];
    }
  }
}

// ---------------------------------------------------------------- RMSNorm + RoPE
// outscale folds the attention softmax 1/sqrt(D) into Q.

__global__ __launch_bounds__(256) void qk_norm_rope(
    const float* __restrict__ proj, int ld, const float* __restrict__ w,
    bf16* __restrict__ out, int nheads, int seq, int posoff, float outscale) {
  const int lane = threadIdx.x & 31;
  const int id   = blockIdx.x * 8 + (threadIdx.x >> 5);
  const int tok  = id / nheads;
  const int h    = id % nheads;

  const float* x = proj + (long)tok * ld + h * HD;
  float v0 = x[lane], v1 = x[lane + 32], v2 = x[lane + 64], v3 = x[lane + 96];
  float ss = wave_sum32(v0 * v0 + v1 * v1 + v2 * v2 + v3 * v3);
  float rn = rsqrtf(ss * (1.0f / HD) + 1e-6f);
  v0 *= rn * w[lane];       v1 *= rn * w[lane + 32];
  v2 *= rn * w[lane + 64];  v3 *= rn * w[lane + 96];

  const float pos = (float)(posoff + tok);
  const float kln = 13.815511f / 64.0f;  // ln(1e6)/64
  float c0, s0, c1, s1;
  __sincosf(pos * __expf(-kln * (float)lane), &s0, &c0);
  __sincosf(pos * __expf(-kln * (float)(lane + 32)), &s1, &c1);

  bf16* o = out + ((long)h * seq + tok) * HD;
  o[lane]      = (bf16)((v0 * c0 - v2 * s0) * outscale);
  o[lane + 32] = (bf16)((v1 * c1 - v3 * s1) * outscale);
  o[lane + 64] = (bf16)((v2 * c0 + v0 * s0) * outscale);
  o[lane + 96] = (bf16)((v3 * c1 + v1 * s1) * outscale);
}

__global__ void v_transpose_cast(const float* __restrict__ kvproj,
                                 bf16* __restrict__ vtb) {
  long n = (long)NKVH * HD * KVL;
  long i = (long)blockIdx.x * blockDim.x + threadIdx.x;
  long stride = (long)gridDim.x * blockDim.x;
  for (; i < n; i += stride) {
    long hd = i / KVL;
    long ki = i - hd * KVL;
    vtb[i] = (bf16)kvproj[ki * 2048 + 1024 + hd];
  }
}

// ---------------------------------------------------------------- flash attention
// Grid (QLEN/128, NH); 8 waves/block share one head and a 128-row q tile.
// Block-uniform kv loop to the tile's causal extent; K/V tiles double-buffered
// in LDS via async-to-LDS; per-row masking handles the diagonal.

#define KST 136  // K tile LDS row stride (272B, 16B aligned): [kv 32][d 128]
#define VST 40   // V tile LDS row stride (80B):               [d 128][kv 32]

__global__ __launch_bounds__(256) void attn_fused(
    const bf16* __restrict__ qbf,   // [NH][QLEN][HD], pre-scaled by 1/sqrt(D)
    const bf16* __restrict__ kbf,   // [NKVH][KVL][HD]
    const bf16* __restrict__ vtb,   // [NKVH][HD][KVL]
    bf16* __restrict__ attnb) {     // [QLEN][NH*HD]
  __shared__ __attribute__((aligned(16))) bf16 Ks[2][32 * KST];
  __shared__ __attribute__((aligned(16))) bf16 Vs[2][128 * VST];
  __shared__ __attribute__((aligned(16))) bf16 P[8][16 * 32];

  const int t    = threadIdx.x;
  const int lane = t & 31;
  const int wave = t >> 5;
  const int g    = lane >> 4;
  const int ln   = lane & 15;
  const int h    = blockIdx.y;
  const int kvh  = h / GRP;
  const int qblk  = blockIdx.x * 128;
  const int qbase = qblk + wave * 16;
  bf16* Pw = P[wave];

  // staging coordinates
  const int krow  = t >> 3;          // 0..31  (kv row), 8 threads/row
  const int kdoff = (t & 7) * 16;    // d offset, 16 bf16 per thread
  const int vrow  = t >> 1;          // 0..127 (d row), 2 threads/row
  const int vkoff = (t & 1) * 16;    // kv offset, 16 bf16 per thread
  const bf16* kgbase = kbf + (size_t)kvh * KVL * HD;
  const bf16* vgbase = vtb + (size_t)kvh * HD * KVL;

  auto stage = [&](int sb, int j0) {
    const bf16* gk = kgbase + (size_t)(j0 + krow) * HD + kdoff;
    bf16* lk = &Ks[sb][krow * KST + kdoff];
    cp16(gk, lk);
    cp16(gk + 8, lk + 8);
    const bf16* gv = vgbase + (size_t)vrow * KVL + j0 + vkoff;
    bf16* lv = &Vs[sb][vrow * VST + vkoff];
    cp16(gv, lv);
    cp16(gv + 8, lv + 8);
  };

  // Q fragments: 16 rows x 128 = 4 A-fragments, resident in registers
  v16bf qf[4];
  const bf16* qrow = qbf + ((long)h * QLEN + qbase + ln) * HD;
#pragma unroll
  for (int c = 0; c < 4; ++c) qf[c] = frag_a_row(qrow + 32 * c, g);

  v8f acc[8];
#pragma unroll
  for (int tt = 0; tt < 8; ++tt) acc[tt] = (v8f){0, 0, 0, 0, 0, 0, 0, 0};
  float m[8], l[8];
#pragma unroll
  for (int r = 0; r < 8; ++r) { m[r] = -1e30f; l[r] = 0.0f; }

  const int kv_end = CTXL + qblk + 128;  // block-uniform causal extent

  stage(0, 0);
  int buf = 0;
  for (int j0 = 0; j0 < kv_end; j0 += 32) {
    wait_stage();
    __syncthreads();
    if (j0 + 32 < kv_end) stage(buf ^ 1, j0 + 32);

    // --- S = Q * K^T for 32 kv columns (two 16-col C-tiles)
    v8f S0 = (v8f){0, 0, 0, 0, 0, 0, 0, 0};
    v8f S1 = (v8f){0, 0, 0, 0, 0, 0, 0, 0};
#pragma unroll
    for (int c = 0; c < 4; ++c) {
      S0 = wmma_bf16(qf[c], frag_b16(&Ks[buf][ln * KST + 32 * c + g * 16]), S0);
      S1 = wmma_bf16(qf[c], frag_b16(&Ks[buf][(16 + ln) * KST + 32 * c + g * 16]), S1);
    }

    // --- masked online softmax, row M = r + 8g, col N = ln
#pragma unroll
    for (int r = 0; r < 8; ++r) {
      const int qpos = CTXL + qbase + r + 8 * g;
      float s0 = S0[r], s1 = S1[r];
      if (j0 + ln > qpos)      s0 = -1e30f;
      if (j0 + 16 + ln > qpos) s1 = -1e30f;
      float mn = fmaxf(m[r], half_max16(fmaxf(s0, s1)));
      float f  = __expf(m[r] - mn);
      float p0 = __expf(s0 - mn);
      float p1 = __expf(s1 - mn);
      l[r] = l[r] * f + half_sum16(p0 + p1);
      m[r] = mn;
#pragma unroll
      for (int tt = 0; tt < 8; ++tt) acc[tt][r] *= f;
      Pw[(r + 8 * g) * 32 + ln]      = (bf16)p0;
      Pw[(r + 8 * g) * 32 + 16 + ln] = (bf16)p1;
    }

    // --- O += P(16x32) * V(32x128)
    v16bf pf = frag_a_row(&Pw[ln * 32], g);
#pragma unroll
    for (int tt = 0; tt < 8; ++tt)
      acc[tt] = wmma_bf16(pf, frag_b16(&Vs[buf][(tt * 16 + ln) * VST + g * 16]),
                          acc[tt]);

    __syncthreads();
    buf ^= 1;
  }

#pragma unroll
  for (int r = 0; r < 8; ++r) {
    float inv = 1.0f / l[r];
    long row = (long)(qbase + r + 8 * g) * HID + h * HD;
#pragma unroll
    for (int tt = 0; tt < 8; ++tt)
      attnb[row + tt * 16 + ln] = (bf16)(acc[tt][r] * inv);
  }
}

// ---------------------------------------------------------------- launch

extern "C" void kernel_launch(void* const* d_in, const int* in_sizes, int n_in,
                              void* d_out, int out_size, void* d_ws, size_t ws_size,
                              hipStream_t stream) {
  const float* context  = (const float*)d_in[0];  // [4096,4096]
  const float* query    = (const float*)d_in[1];  // [2048,4096]
  const float* w_qkv    = (const float*)d_in[2];  // [4096,6144]
  const float* w_o      = (const float*)d_in[3];  // [4096,4096]
  const float* q_norm_w = (const float*)d_in[4];  // [128]
  const float* k_norm_w = (const float*)d_in[5];  // [128]
  float* out = (float*)d_out;
  (void)in_sizes; (void)n_in; (void)out_size; (void)ws_size;

  char* base = (char*)d_ws;
  size_t off = 0;
  auto carve = [&](size_t bytes) -> char* {
    char* r = base + off;
    off += (bytes + 255) & ~(size_t)255;
    return r;
  };
  bf16*  ckb    = (bf16*)carve((size_t)KVL * HID * 2);
  bf16*  wqkvT  = (bf16*)carve((size_t)QKVC * HID * 2);
  bf16*  woT    = (bf16*)carve((size_t)HID * HID * 2);
  float* qproj  = (float*)carve((size_t)QLEN * HID * 4);
  float* kvproj = (float*)carve((size_t)KVL * 2048 * 4);
  bf16*  qbf    = (bf16*)carve((size_t)NH * QLEN * HD * 2);
  bf16*  kbf    = (bf16*)carve((size_t)NKVH * KVL * HD * 2);
  bf16*  vtb    = (bf16*)carve((size_t)NKVH * HD * KVL * 2);
  bf16*  attnb  = (bf16*)carve((size_t)QLEN * HID * 2);

  cast_f32_bf16<<<4096, 256, 0, stream>>>(ckb, context, (long)CTXL * HID);
  cast_f32_bf16<<<4096, 256, 0, stream>>>(ckb + (size_t)CTXL * HID, query,
                                          (long)QLEN * HID);
  cast_transpose_f32_bf16<<<dim3(QKVC / 32, HID / 32), 256, 0, stream>>>(
      wqkvT, w_qkv, HID, QKVC);
  cast_transpose_f32_bf16<<<dim3(HID / 32, HID / 32), 256, 0, stream>>>(
      woT, w_o, HID, HID);

  gemm_bf16f32<<<dim3(HID / 256, QLEN / 128), 256, 0, stream>>>(
      ckb + (size_t)CTXL * HID, HID, wqkvT, HID, qproj, HID, HID);
  gemm_bf16f32<<<dim3(2048 / 256, KVL / 128), 256, 0, stream>>>(
      ckb, HID, wqkvT + (size_t)(NH * HD) * HID, HID, kvproj, 2048, HID);

  qk_norm_rope<<<(QLEN * NH) / 8, 256, 0, stream>>>(qproj, HID, q_norm_w, qbf,
                                                    NH, QLEN, CTXL,
                                                    0.0883883476f);
  qk_norm_rope<<<(KVL * NKVH) / 8, 256, 0, stream>>>(kvproj, 2048, k_norm_w, kbf,
                                                     NKVH, KVL, 0, 1.0f);
  v_transpose_cast<<<8192, 256, 0, stream>>>(kvproj, vtb);

  attn_fused<<<dim3(QLEN / 128, NH), 256, 0, stream>>>(qbf, kbf, vtb, attnb);

  gemm_bf16f32<<<dim3(HID / 256, QLEN / 128), 256, 0, stream>>>(
      attnb, HID, woT, HID, out, HID, HID);
}
